// SceneSegmenterModel_4105988735324
// MI455X (gfx1250) — compile-verified, matrix operand order checked by reading, not run on hardware
//
#include <hip/hip_runtime.h>
#include <hip/hip_bf16.h>

// ---------------------------------------------------------------------------
// CDNA5 / gfx1250 implementation. wave32 only. All matmuls use
// v_wmma_f32_16x16x32_f16 (D = A(16x32,f16) x B(32x16,f16) + C(16x16,f32)).
// Activations kept f16 in workspace; softmax stats and epilogues in f32.
// Tile staging uses GLOBAL_LOAD_ASYNC_TO_LDS_B128 (ASYNCcnt) where no format
// conversion is needed.
// ---------------------------------------------------------------------------

typedef __attribute__((ext_vector_type(16))) _Float16 v16h;
typedef __attribute__((ext_vector_type(8)))  _Float16 v8h;
typedef __attribute__((ext_vector_type(8)))  float    v8f;
typedef int vint4 __attribute__((vector_size(16)));   // matches async builtin param

union F16Frag { v16h v; v8h h[2]; };

#if __has_builtin(__builtin_amdgcn_global_load_async_to_lds_b128) && \
    __has_builtin(__builtin_amdgcn_s_wait_asynccnt)
#define USE_ASYNC_LDS 1
#else
#define USE_ASYNC_LDS 0
#endif

#define ASG __attribute__((address_space(1)))
#define ASL __attribute__((address_space(3)))

__device__ __forceinline__ v8f wmma16x16x32(v16h a, v16h b, v8f c) {
    return __builtin_amdgcn_wmma_f32_16x16x32_f16(
        /*neg_a=*/false, a, /*neg_b=*/false, b,
        /*c_mod=*/(short)0, c, /*reuse_a=*/false, /*reuse_b=*/false);
}

// Load one 16x32 f16 operand (A layout; B uses the same layout applied to
// the (N,K) matrix) from a row-major f16 tile with `stride` halves per row.
// Per-lane packing per ISA 7.12.2:
//   lanes 0-15 : K = 0..7 (elems 0..7),  K = 16..23 (elems 8..15)
//   lanes 16-31: K = 8..15(elems 0..7),  K = 24..31 (elems 8..15)
// `stride*2` bytes must be a multiple of 16 for aligned b128 LDS reads.
__device__ __forceinline__ v16h load_frag(const _Float16* base, int stride) {
    const int lane = threadIdx.x & 31;
    const int r    = lane & 15;
    const int k0   = (lane >> 4) << 3;   // 0 or 8
    const _Float16* rp = base + r * stride;
    F16Frag f;
    f.h[0] = *(const v8h*)(rp + k0);
    f.h[1] = *(const v8h*)(rp + k0 + 16);
    return f.v;
}

// ---------------------------------------------------------------------------
// Generic GEMM: C[M,N] = act(A[M,K] * W[N,K]^T + bias[N])
//   A: f32 or f16 (row-major, leading dim lda, column offset a_off)
//   W: f32 (N,K) row-major, converted to f16 in LDS per tile
//   C: f32 or f16, ldc leading dim, column offset c_off
//   PERM: output row remap token(b*permS+s) -> s*permB + b (seq-first store)
// Block tile 64x32, K-step 32, 8 waves (4x2 grid of 16x16 WMMA tiles).
// ---------------------------------------------------------------------------
#define GBM 64
#define GBN 32
#define GBK 32
#define LDSP 40   // padded LDS row stride in halves (80B, 16B-aligned)

template<bool A_F16, bool OUT_F16, bool RELU, bool PERM>
__global__ __launch_bounds__(256) void gemm_kernel(
        const void* __restrict__ Ap, long lda, long a_off,
        const float* __restrict__ W, const float* __restrict__ bias,
        void* __restrict__ Cp, long ldc, long c_off,
        int M, int N, int K, int permS, int permB)
{
    __shared__ _Float16 As[GBM * LDSP];
    __shared__ _Float16 Ws[GBN * LDSP];

    const int tid  = threadIdx.x;
    const int lane = tid & 31;
    const int wave = tid >> 5;
    const int mi   = wave & 3;     // 0..3  (16-row sub-tile)
    const int ni   = wave >> 2;    // 0..1  (16-col sub-tile)
    const long m_base = (long)blockIdx.x * GBM;
    const long n_base = (long)blockIdx.y * GBN;

    v8f acc = {0.f, 0.f, 0.f, 0.f, 0.f, 0.f, 0.f, 0.f};

    for (int k0 = 0; k0 < K; k0 += GBK) {
        __syncthreads();
        // ---- stage A tile (64x32) into LDS as f16 ----
        {
            const int r = tid >> 2;            // 0..63
            const int c = (tid & 3) * 8;       // 0,8,16,24
            long gr = m_base + r; if (gr >= M) gr = M - 1;
            if constexpr (A_F16) {
                const _Float16* A = (const _Float16*)Ap;
                const _Float16* src = A + gr * lda + a_off + k0 + c;
#if USE_ASYNC_LDS
                // async direct memory->LDS copy (16B), ASYNCcnt-tracked
                __builtin_amdgcn_global_load_async_to_lds_b128(
                    (ASG vint4*)src, (ASL vint4*)&As[r * LDSP + c], 0, 0);
#else
                *(v8h*)&As[r * LDSP + c] = *(const v8h*)src;
#endif
                if (k0 + GBK < K) __builtin_prefetch((const void*)(src + GBK), 0, 1);
            } else {
                const float* A = (const float*)Ap;
                const float* src = A + gr * lda + a_off + k0 + c;
                v8h vals;
                #pragma unroll
                for (int j = 0; j < 8; ++j) vals[j] = (_Float16)src[j];
                *(v8h*)&As[r * LDSP + c] = vals;
                if (k0 + GBK < K) __builtin_prefetch((const void*)(src + GBK), 0, 1);
            }
        }
        // ---- stage W tile (32x32, N x K) into LDS as f16 (f32->f16) ----
        {
            const int r = tid >> 3;            // 0..31
            const int c = (tid & 7) * 4;       // 0..28
            long gn = n_base + r; if (gn >= N) gn = N - 1;
            const float* src = W + gn * (long)K + k0 + c;
            #pragma unroll
            for (int j = 0; j < 4; ++j) Ws[r * LDSP + c + j] = (_Float16)src[j];
        }
#if USE_ASYNC_LDS
        if constexpr (A_F16) __builtin_amdgcn_s_wait_asynccnt(0);
#endif
        __syncthreads();

        v16h a = load_frag(As + mi * 16 * LDSP, LDSP);
        v16h b = load_frag(Ws + ni * 16 * LDSP, LDSP);
        acc = wmma16x16x32(a, b, acc);
    }

    // ---- epilogue: bias + act + store ----
    const int r16 = lane & 15;
    const int hi  = lane >> 4;
    const long n_local = n_base + ni * 16 + r16;
    const float bval = bias[n_local];
    #pragma unroll
    for (int r = 0; r < 8; ++r) {
        const long gm = m_base + mi * 16 + r + 8 * hi;
        if (gm < M && n_local < N) {
            float v = acc[r] + bval;
            if constexpr (RELU) v = fmaxf(v, 0.f);
            long orow = gm;
            if constexpr (PERM) orow = (gm % permS) * (long)permB + (gm / permS);
            if constexpr (OUT_F16)
                ((_Float16*)Cp)[orow * ldc + c_off + n_local] = (_Float16)v;
            else
                ((float*)Cp)[orow * ldc + c_off + n_local] = v;
        }
    }
}

// ---------------------------------------------------------------------------
// Flash attention, single head, D=128, scale 1/sqrt(128).
// qkv: (B, S, 384) f16 with q|k|v at col offsets 0|128|256 (batch = blockIdx.y)
// O  : (B, S, 128) f16
// Block = 4 waves; each wave owns 16 query rows; block streams 32-key chunks
// of K (row-major, async-copied) and V (transposed at staging time) through
// LDS shared by the 4 waves. Online (streaming) softmax in f32.
// ---------------------------------------------------------------------------
#define KVP 136   // LDS stride for K chunk (272B rows, 16B-aligned)

__global__ __launch_bounds__(128) void attn_kernel(
        const _Float16* __restrict__ qkv, _Float16* __restrict__ O, int S)
{
    __shared__ _Float16 kbuf[32 * KVP];      // 32 keys x 128 d (row-major)
    __shared__ _Float16 vbufT[128 * LDSP];   // 128 n   x 32 keys (transposed)
    __shared__ _Float16 pbuf[4][16 * LDSP];  // per-wave P (16 x 32)

    const int lane  = threadIdx.x & 31;
    const int wave  = threadIdx.x >> 5;
    const int r16   = lane & 15;
    const int hi    = lane >> 4;
    const int k0    = hi << 3;
    const long bOff = (long)blockIdx.y * S * 384;
    const long qbase = (long)blockIdx.x * 64 + wave * 16;

    // ---- load this wave's Q fragments (16 x 128 = 4 x (16x32)) ----
    v16h aq[4];
    {
        long qr = qbase + r16; if (qr >= S) qr = S - 1;
        const _Float16* qp = qkv + bOff + qr * 384;
        #pragma unroll
        for (int kk = 0; kk < 4; ++kk) {
            F16Frag f;
            f.h[0] = *(const v8h*)(qp + kk * 32 + k0);
            f.h[1] = *(const v8h*)(qp + kk * 32 + k0 + 16);
            aq[kk] = f.v;
        }
    }

    float mrow[8], lrow[8];
    v8f accO[8];
    #pragma unroll
    for (int r = 0; r < 8; ++r) { mrow[r] = -1e30f; lrow[r] = 0.f; }
    #pragma unroll
    for (int t = 0; t < 8; ++t)
        accO[t] = (v8f){0.f, 0.f, 0.f, 0.f, 0.f, 0.f, 0.f, 0.f};

    const float scale = 0.08838834764831845f;   // 1/sqrt(128)

    for (int kc = 0; kc < S; kc += 32) {
        __syncthreads();
        // ---- cooperative stage of K (async, row-major) and V (transposed) ----
        {
            const int key = threadIdx.x >> 2;          // 0..31
            const int c   = (threadIdx.x & 3) * 32;    // 0,32,64,96
            const long gk = kc + key;
            if (gk < S) {
                const _Float16* kp = qkv + bOff + gk * 384 + 128 + c;
                const _Float16* vp = qkv + bOff + gk * 384 + 256 + c;
                #pragma unroll
                for (int j = 0; j < 4; ++j) {
#if USE_ASYNC_LDS
                    __builtin_amdgcn_global_load_async_to_lds_b128(
                        (ASG vint4*)(kp + 8 * j),
                        (ASL vint4*)&kbuf[key * KVP + c + 8 * j], 0, 0);
#else
                    *(v8h*)&kbuf[key * KVP + c + 8 * j] = *(const v8h*)(kp + 8 * j);
#endif
                    const v8h vv = *(const v8h*)(vp + 8 * j);
                    #pragma unroll
                    for (int u = 0; u < 8; ++u)
                        vbufT[(c + 8 * j + u) * LDSP + key] = vv[u];
                }
            } else {
                const v8h z = {};
                #pragma unroll
                for (int j = 0; j < 4; ++j) {
                    *(v8h*)&kbuf[key * KVP + c + 8 * j] = z;
                    #pragma unroll
                    for (int u = 0; u < 8; ++u)
                        vbufT[(c + 8 * j + u) * LDSP + key] = (_Float16)0.f;
                }
            }
        }
#if USE_ASYNC_LDS
        __builtin_amdgcn_s_wait_asynccnt(0);
#endif
        __syncthreads();

        // ---- scores: (16 q) x (32 keys), contraction over d=128 ----
        v8f s0 = {0.f,0.f,0.f,0.f,0.f,0.f,0.f,0.f};
        v8f s1 = {0.f,0.f,0.f,0.f,0.f,0.f,0.f,0.f};
        #pragma unroll
        for (int kk = 0; kk < 4; ++kk) {
            v16h bk0 = load_frag(kbuf + kk * 32, KVP);            // keys 0..15
            v16h bk1 = load_frag(kbuf + 16 * KVP + kk * 32, KVP); // keys 16..31
            s0 = wmma16x16x32(aq[kk], bk0, s0);
            s1 = wmma16x16x32(aq[kk], bk1, s1);
        }
        const bool ok0 = (kc + r16) < S;
        const bool ok1 = (kc + 16 + r16) < S;
        #pragma unroll
        for (int r = 0; r < 8; ++r) {
            s0[r] = ok0 ? s0[r] * scale : -1e30f;
            s1[r] = ok1 ? s1[r] * scale : -1e30f;
        }

        // ---- online softmax (per row; rows live across 16-lane halves) ----
        #pragma unroll
        for (int r = 0; r < 8; ++r) {
            float mx = fmaxf(s0[r], s1[r]);
            #pragma unroll
            for (int off = 8; off > 0; off >>= 1)
                mx = fmaxf(mx, __shfl_xor(mx, off, 32));
            const float mn   = fmaxf(mrow[r], mx);
            const float corr = __expf(mrow[r] - mn);
            const float p0   = __expf(s0[r] - mn);
            const float p1   = __expf(s1[r] - mn);
            float rs = p0 + p1;
            #pragma unroll
            for (int off = 8; off > 0; off >>= 1)
                rs += __shfl_xor(rs, off, 32);
            lrow[r] = lrow[r] * corr + rs;
            mrow[r] = mn;
            #pragma unroll
            for (int t = 0; t < 8; ++t) accO[t][r] = accO[t][r] * corr;
            const int m = r + 8 * hi;
            pbuf[wave][m * LDSP + r16]      = (_Float16)p0;
            pbuf[wave][m * LDSP + 16 + r16] = (_Float16)p1;
        }

        // ---- accO += P(16x32) x V(32x128); V already transposed in LDS ----
        v16h pf = load_frag(pbuf[wave], LDSP);
        #pragma unroll
        for (int t = 0; t < 8; ++t) {
            v16h bv = load_frag(vbufT + (t * 16) * LDSP, LDSP);
            accO[t] = wmma16x16x32(pf, bv, accO[t]);
        }
    }

    // ---- normalize and store O (f16) ----
    _Float16* Ob = O + (long)blockIdx.y * S * 128;
    #pragma unroll
    for (int r = 0; r < 8; ++r) {
        const long row = qbase + r + 8 * hi;
        if (row < S) {
            const float inv = 1.f / lrow[r];
            #pragma unroll
            for (int t = 0; t < 8; ++t)
                Ob[row * 128 + t * 16 + r16] = (_Float16)(accO[t][r] * inv);
        }
    }
}

// ---------------------------------------------------------------------------
// Segmentation prefix/suffix means.  e: (S,B,128) f32 (= d_out region).
// cat: (B, S-1, 256) f16, [0:128)=left mean, [128:256)=right mean.
// One thread per (b,d) column walks S elements twice (1024 threads total).
// ---------------------------------------------------------------------------
__global__ void seg_kernel(const float* __restrict__ e,
                           _Float16* __restrict__ cat, int S, int B)
{
    const int idx = blockIdx.x * blockDim.x + threadIdx.x;
    if (idx >= B * 128) return;
    const int b = idx >> 7, d = idx & 127;
    const float* p = e + b * 128 + d;
    const long stride = (long)B * 128;
    float total = 0.f;
    for (int s = 0; s < S; ++s) total += p[s * stride];
    float csum = 0.f;
    for (int j = 0; j < S - 1; ++j) {
        csum += p[j * stride];
        const float left  = csum / (float)(j + 1);
        const float right = (total - csum) / (float)(S - 1 - j);
        const long row = (long)b * (S - 1) + j;
        cat[row * 256 + d]       = (_Float16)left;
        cat[row * 256 + 128 + d] = (_Float16)right;
    }
}

// ---------------------------------------------------------------------------
// Final boundary head: sigmoid(o2p[token,:] . bp_W + bp_b) -> out[s*B + b]
// o2p rows are token = b*(S-1) + s.
// ---------------------------------------------------------------------------
__global__ void bp_kernel(const _Float16* __restrict__ o2p,
                          const float* __restrict__ bpW,
                          const float* __restrict__ bpb,
                          float* __restrict__ out, int S1, int B)
{
    const int token = blockIdx.x * blockDim.x + threadIdx.x;
    if (token >= S1 * B) return;
    const int b = token / S1, s = token % S1;
    const _Float16* row = o2p + (long)token * 128;
    float acc = bpb[0];
    #pragma unroll
    for (int d = 0; d < 128; d += 8) {
        const v8h rv = *(const v8h*)(row + d);
        #pragma unroll
        for (int u = 0; u < 8; ++u) acc += (float)rv[u] * bpW[d + u];
    }
    out[(long)s * B + b] = 1.f / (1.f + __expf(-acc));
}

// ---------------------------------------------------------------------------
extern "C" void kernel_launch(void* const* d_in, const int* in_sizes, int n_in,
                              void* d_out, int out_size, void* d_ws, size_t ws_size,
                              hipStream_t stream) {
    (void)in_sizes; (void)n_in; (void)out_size; (void)ws_size;
    constexpr int  B = 8, S = 2048, F = 3584, D = 128;
    constexpr long T  = (long)B * S;          // 16384
    constexpr int  S1 = S - 1;                // 2047
    constexpr long T2 = (long)B * S1;         // 16376
    constexpr long E_ELEMS = (long)S * B * D; // 2097152

    const float* x       = (const float*)d_in[0];
    const float* place_W = (const float*)d_in[1];  const float* place_b = (const float*)d_in[2];
    const float* cast_W  = (const float*)d_in[3];  const float* cast_b  = (const float*)d_in[4];
    const float* act_W   = (const float*)d_in[5];  const float* act_b   = (const float*)d_in[6];
    const float* aud_W   = (const float*)d_in[7];  const float* aud_b   = (const float*)d_in[8];
    const float* e_W     = (const float*)d_in[9];  const float* e_b     = (const float*)d_in[10];
    const float* la_win  = (const float*)d_in[11]; const float* la_bin  = (const float*)d_in[12];
    const float* la_wout = (const float*)d_in[13]; const float* la_bout = (const float*)d_in[14];
    const float* be_W    = (const float*)d_in[15]; const float* be_b    = (const float*)d_in[16];
    const float* ba_win  = (const float*)d_in[17]; const float* ba_bin  = (const float*)d_in[18];
    const float* ba_wout = (const float*)d_in[19]; const float* ba_bout = (const float*)d_in[20];
    const float* bp_W    = (const float*)d_in[21]; const float* bp_b    = (const float*)d_in[22];
    float* out = (float*)d_out;

    char* ws = (char*)d_ws;
    _Float16* feats = (_Float16*)(ws);                                   // T x 512
    _Float16* e0    = (_Float16*)(ws + 16777216);                        // T x 128
    _Float16* qkv1  = (_Float16*)(ws + 20971520);                        // T x 384
    _Float16* o1    = (_Float16*)(ws + 33554432);                        // T x 128
    _Float16* cat   = (_Float16*)(ws + 37748736);                        // B x S1 x 256
    _Float16* b0    = (_Float16*)(ws + 46133248);                        // T2 x 128
    _Float16* qkv2  = (_Float16*)(ws + 50325504);                        // T2 x 384
    _Float16* o2    = (_Float16*)(ws + 62902272);                        // T2 x 128
    _Float16* o2p   = (_Float16*)(ws + 67094528);                        // T2 x 128

    const dim3 blk256(256), blk128(128);
    const dim3 gT4((T + GBM - 1) / GBM, 4);     // M=T,  N=128
    const dim3 gT12((T + GBM - 1) / GBM, 12);   // M=T,  N=384
    const dim3 gT24((T2 + GBM - 1) / GBM, 4);   // M=T2, N=128
    const dim3 gT212((T2 + GBM - 1) / GBM, 12); // M=T2, N=384

    // 1) slice projections -> feats (f16, T x 512)
    gemm_kernel<false,true,true,false><<<gT4, blk256, 0, stream>>>(
        x, F, 0,    place_W, place_b, feats, 512, 0,   (int)T, 128, 2048, 0, 0);
    gemm_kernel<false,true,true,false><<<gT4, blk256, 0, stream>>>(
        x, F, 2048, cast_W,  cast_b,  feats, 512, 128, (int)T, 128, 512,  0, 0);
    gemm_kernel<false,true,true,false><<<gT4, blk256, 0, stream>>>(
        x, F, 2560, act_W,   act_b,   feats, 512, 256, (int)T, 128, 512,  0, 0);
    gemm_kernel<false,true,true,false><<<gT4, blk256, 0, stream>>>(
        x, F, 3072, aud_W,   aud_b,   feats, 512, 384, (int)T, 128, 512,  0, 0);
    // 2) e0 = relu(feats @ e_W^T + e_b)
    gemm_kernel<true,true,true,false><<<gT4, blk256, 0, stream>>>(
        feats, 512, 0, e_W, e_b, e0, 128, 0, (int)T, 128, 512, 0, 0);
    // 3) qkv projection (MHA1)
    gemm_kernel<true,true,false,false><<<gT12, blk256, 0, stream>>>(
        e0, 128, 0, la_win, la_bin, qkv1, 384, 0, (int)T, 384, 128, 0, 0);
    // 4) attention 1
    attn_kernel<<<dim3((S + 63) / 64, B), blk128, 0, stream>>>(qkv1, o1, S);
    // 5) out projection -> d_out e region, permuted to (S,B,D)
    gemm_kernel<true,false,false,true><<<gT4, blk256, 0, stream>>>(
        o1, 128, 0, la_wout, la_bout, out, 128, 0, (int)T, 128, 128, S, B);
    // 6) prefix/suffix means -> cat
    seg_kernel<<<dim3((B * 128 + 255) / 256), blk256, 0, stream>>>(out, cat, S, B);
    // 7) boundary embedding
    gemm_kernel<true,true,true,false><<<gT24, blk256, 0, stream>>>(
        cat, 256, 0, be_W, be_b, b0, 128, 0, (int)T2, 128, 256, 0, 0);
    // 8) qkv projection (MHA2)
    gemm_kernel<true,true,false,false><<<gT212, blk256, 0, stream>>>(
        b0, 128, 0, ba_win, ba_bin, qkv2, 384, 0, (int)T2, 384, 128, 0, 0);
    // 9) attention 2 (S=2047, tail masked)
    attn_kernel<<<dim3((S1 + 63) / 64, B), blk128, 0, stream>>>(qkv2, o2, S1);
    // 10) out projection 2
    gemm_kernel<true,true,false,false><<<gT24, blk256, 0, stream>>>(
        o2, 128, 0, ba_wout, ba_bout, o2p, 128, 0, (int)T2, 128, 128, 0, 0);
    // 11) boundary head -> d_out b region (seq-major)
    bp_kernel<<<dim3((int)((T2 + 255) / 256)), blk256, 0, stream>>>(
        o2p, bp_W, bp_b, out + E_ELEMS, S1, B);
}
